// Pointnet_90752658965070
// MI455X (gfx1250) — matrix operand order repaired and checked by reading, not compiled, not run on hardware
//
#include <hip/hip_runtime.h>
#include <hip/hip_bf16.h>
#include <stdint.h>

// PointNet++ MSG forward for MI455X (gfx1250, wave32).
// Channel-mixing convs -> v_wmma_f32_16x16x32_f16 GEMMs with 2x2 register
// blocking (one wave = 32x32 output tile, 4 accumulators), f16 activations,
// two-pass BN (stats pass + apply pass), fused ReLU + max-pool via atomicMax.

typedef _Float16 f16;
typedef __attribute__((ext_vector_type(16))) _Float16 v16h;
typedef __attribute__((ext_vector_type(8)))  float    v8f;

#define BB   16
#define NPTS 8192

union V16 { v16h v; uint4 q[2]; };

// ---------------------------------------------------------------------------
// WMMA 32x32 block (2x2 tiles) of Y = X (MxKpad f16 row-major) * W^T (CoutxKpad)
// ---------------------------------------------------------------------------
struct Acc4 { v8f a00, a01, a10, a11; };

__device__ __forceinline__ Acc4 wmma_block(const f16* __restrict__ X,
                                           const f16* __restrict__ W,
                                           int Kpad, int tm2, int tn2)
{
    const int lane = threadIdx.x & 31;
    const int r    = lane & 15;
    const int half = lane >> 4;
    const f16* x0 = X + (size_t)(tm2 * 32 + r) * Kpad;
    const f16* x1 = x0 + (size_t)16 * Kpad;
    const f16* w0 = W + (size_t)(tn2 * 32 + r) * Kpad;
    const f16* w1 = w0 + (size_t)16 * Kpad;
    Acc4 acc;
    acc.a00 = {}; acc.a01 = {}; acc.a10 = {}; acc.a11 = {};
    for (int k0 = 0; k0 < Kpad; k0 += 32) {
        V16 a0, a1, b0, b1;
        // A 16x32 f16: elems 0..7 = K[8h..8h+7], elems 8..15 = K[16+8h..23+8h]
        a0.q[0] = *(const uint4*)(x0 + k0 + 8 * half);
        a0.q[1] = *(const uint4*)(x0 + k0 + 16 + 8 * half);
        a1.q[0] = *(const uint4*)(x1 + k0 + 8 * half);
        a1.q[1] = *(const uint4*)(x1 + k0 + 16 + 8 * half);
        // B 32x16 f16 (fed as B^T rows): elems 0..7 = K[8h..8h+7], elems 8..15 = K[16+8h..23+8h]
        b0.q[0] = *(const uint4*)(w0 + k0 + 8 * half);
        b0.q[1] = *(const uint4*)(w0 + k0 + 16 + 8 * half);
        b1.q[0] = *(const uint4*)(w1 + k0 + 8 * half);
        b1.q[1] = *(const uint4*)(w1 + k0 + 16 + 8 * half);
        __builtin_prefetch(x0 + k0 + 64, 0, 1);
        __builtin_prefetch(x1 + k0 + 64, 0, 1);
        acc.a00 = __builtin_amdgcn_wmma_f32_16x16x32_f16(false, a0.v, false, b0.v,
                                                         (short)0, acc.a00, false, false);
        acc.a01 = __builtin_amdgcn_wmma_f32_16x16x32_f16(false, a0.v, false, b1.v,
                                                         (short)0, acc.a01, false, false);
        acc.a10 = __builtin_amdgcn_wmma_f32_16x16x32_f16(false, a1.v, false, b0.v,
                                                         (short)0, acc.a10, false, false);
        acc.a11 = __builtin_amdgcn_wmma_f32_16x16x32_f16(false, a1.v, false, b1.v,
                                                         (short)0, acc.a11, false, false);
    }
    return acc;
}

// D layout (16x16 f32): lane l -> column l&15, rows 8*(l>>4) + r (r = vgpr idx)
__device__ __forceinline__ void stats_tile(const v8f& acc, int ng,
                                           float* __restrict__ ssum,
                                           float* __restrict__ ssq)
{
    float s = 0.f, s2 = 0.f;
#pragma unroll
    for (int r = 0; r < 8; ++r) { float v = acc[r]; s += v; s2 += v * v; }
    atomicAdd(&ssum[ng], s);
    atomicAdd(&ssq [ng], s2);
}

__device__ __forceinline__ void apply_store_tile(const v8f& acc, int m0, int ng,
                                                 float sc, float sh,
                                                 f16* __restrict__ xnext, int Cout)
{
#pragma unroll
    for (int r = 0; r < 8; ++r) {
        float v = fmaxf(0.f, acc[r] * sc + sh);
        xnext[(size_t)(m0 + r) * Cout + ng] = (f16)v;
    }
}

__device__ __forceinline__ void apply_pool_tile(const v8f& acc, int m0, int ng,
                                                float sc, float sh,
                                                float* __restrict__ pool,
                                                int lg2G, int Cout)
{
    float cur = -1.f; int curg = -1;
#pragma unroll
    for (int r = 0; r < 8; ++r) {
        float v = fmaxf(0.f, acc[r] * sc + sh);
        int g = (m0 + r) >> lg2G;
        if (g != curg) {
            if (curg >= 0)
                atomicMax((unsigned int*)&pool[(size_t)curg * Cout + ng],
                          __float_as_uint(cur));
            curg = g; cur = v;
        } else cur = fmaxf(cur, v);
    }
    atomicMax((unsigned int*)&pool[(size_t)curg * Cout + ng], __float_as_uint(cur));
}

// Pass 1: GEMM -> per-channel sum / sumsq (BN statistics)
__global__ void k_gemm_stats(const f16* __restrict__ X, const f16* __restrict__ W,
                             int Kpad, int Mt2, int Nt2,
                             float* __restrict__ ssum, float* __restrict__ ssq)
{
    int wid = blockIdx.x * (blockDim.x >> 5) + (threadIdx.x >> 5);
    if (wid >= Mt2 * Nt2) return;                 // wave-uniform
    int tn2 = wid % Nt2, tm2 = wid / Nt2;
    Acc4 acc = wmma_block(X, W, Kpad, tm2, tn2);
    int lane = threadIdx.x & 31;
    int n   = lane & 15;
    int ng0 = tn2 * 32 + n;
    stats_tile(acc.a00, ng0,      ssum, ssq);
    stats_tile(acc.a01, ng0 + 16, ssum, ssq);
    stats_tile(acc.a10, ng0,      ssum, ssq);
    stats_tile(acc.a11, ng0 + 16, ssum, ssq);
}

// Pass 2: GEMM -> BN(scale,shift) -> ReLU -> (f16 store | fused max-pool)
__global__ void k_gemm_apply(const f16* __restrict__ X, const f16* __restrict__ W,
                             int Kpad, int Mt2, int Nt2,
                             const float* __restrict__ scale,
                             const float* __restrict__ shift,
                             f16* __restrict__ xnext, float* __restrict__ pool,
                             int lg2G, int Cout)
{
    int wid = blockIdx.x * (blockDim.x >> 5) + (threadIdx.x >> 5);
    if (wid >= Mt2 * Nt2) return;                 // wave-uniform
    int tn2 = wid % Nt2, tm2 = wid / Nt2;
    Acc4 acc = wmma_block(X, W, Kpad, tm2, tn2);
    int lane = threadIdx.x & 31;
    int n   = lane & 15;
    int mb  = (lane >> 4) * 8;
    int ng0 = tn2 * 32 + n;
    int ng1 = ng0 + 16;
    int m0  = tm2 * 32 + mb;
    int m1  = m0 + 16;
    float sc0 = scale[ng0], sh0 = shift[ng0];
    float sc1 = scale[ng1], sh1 = shift[ng1];
    if (pool) {
        apply_pool_tile(acc.a00, m0, ng0, sc0, sh0, pool, lg2G, Cout);
        apply_pool_tile(acc.a01, m0, ng1, sc1, sh1, pool, lg2G, Cout);
        apply_pool_tile(acc.a10, m1, ng0, sc0, sh0, pool, lg2G, Cout);
        apply_pool_tile(acc.a11, m1, ng1, sc1, sh1, pool, lg2G, Cout);
    } else {
        apply_store_tile(acc.a00, m0, ng0, sc0, sh0, xnext, Cout);
        apply_store_tile(acc.a01, m0, ng1, sc1, sh1, xnext, Cout);
        apply_store_tile(acc.a10, m1, ng0, sc0, sh0, xnext, Cout);
        apply_store_tile(acc.a11, m1, ng1, sc1, sh1, xnext, Cout);
    }
}

// ---------------------------------------------------------------------------
// Support kernels
// ---------------------------------------------------------------------------
__global__ void k_zero_f32(float* p, int n)
{
    int gid = blockIdx.x * blockDim.x + threadIdx.x;
    if (gid < n) p[gid] = 0.f;
}

__global__ void k_convert_w(const float* __restrict__ w, int Cin, int Kpad, int n,
                            f16* __restrict__ out)
{
    int gid = blockIdx.x * blockDim.x + threadIdx.x;
    if (gid >= n) return;
    int o = gid / Kpad, c = gid % Kpad;
    out[gid] = (c < Cin) ? (f16)w[(size_t)o * Cin + c] : (f16)0.f;
}

// stats layout: [0..1023]=sum [1024..2047]=sumsq [2048..3071]=scale [3072..4095]=shift
__global__ void k_bn_finalize(const float* __restrict__ g, const float* __restrict__ be,
                              float* __restrict__ stats, int Cout, float invM)
{
    int c = blockIdx.x * blockDim.x + threadIdx.x;
    if (c >= Cout) return;
    float mean = stats[c] * invM;
    float var  = stats[1024 + c] * invM - mean * mean;
    float inv  = rsqrtf(var + 1e-5f);
    float sc   = g[c] * inv;
    stats[2048 + c] = sc;
    stats[3072 + c] = be[c] - mean * sc;
}

// Farthest point sampling: one block per batch, dist kept in LDS.
__global__ void k_fps(const float* __restrict__ xyz, int N, int npoint,
                      int* __restrict__ idx_out, float* __restrict__ nxyz)
{
    __shared__ float dist[NPTS];
    __shared__ float redv[256];
    __shared__ int   redi[256];
    __shared__ int   s_far;
    int b = blockIdx.x, t = threadIdx.x;
    const float* P = xyz + (size_t)b * N * 3;
    for (int i = t; i < N; i += 256) dist[i] = 1e10f;
    if (t == 0) s_far = 0;
    __syncthreads();
    for (int it = 0; it < npoint; ++it) {
        int far = s_far;
        float cx = P[far * 3], cy = P[far * 3 + 1], cz = P[far * 3 + 2];
        if (t == 0) {
            idx_out[b * npoint + it] = far;
            nxyz[((size_t)b * npoint + it) * 3 + 0] = cx;
            nxyz[((size_t)b * npoint + it) * 3 + 1] = cy;
            nxyz[((size_t)b * npoint + it) * 3 + 2] = cz;
        }
        float bv = -1.f; int bi = 0;
        for (int i = t; i < N; i += 256) {
            float dx = P[i * 3] - cx, dy = P[i * 3 + 1] - cy, dz = P[i * 3 + 2] - cz;
            float d  = dx * dx + dy * dy + dz * dz;
            float nd = fminf(dist[i], d);
            dist[i] = nd;
            if (nd > bv) { bv = nd; bi = i; }
        }
        redv[t] = bv; redi[t] = bi;
        __syncthreads();
        for (int off = 128; off > 0; off >>= 1) {
            if (t < off) {
                if (redv[t + off] > redv[t] ||
                    (redv[t + off] == redv[t] && redi[t + off] < redi[t])) {
                    redv[t] = redv[t + off]; redi[t] = redi[t + off];
                }
            }
            __syncthreads();
        }
        if (t == 0) s_far = redi[0];
        __syncthreads();
    }
}

// Ball query: ordered first-K selection (== reference's sort-based select).
__global__ void k_ball(const float* __restrict__ xyz, const float* __restrict__ nxyz,
                       int B_, int N, int S, int K, float r2, int* __restrict__ idx)
{
    int gid = blockIdx.x * blockDim.x + threadIdx.x;
    if (gid >= B_ * S) return;
    int b = gid / S;
    const float* P = xyz + (size_t)b * N * 3;
    float cx = nxyz[gid * 3], cy = nxyz[gid * 3 + 1], cz = nxyz[gid * 3 + 2];
    int cnt = 0, first = 0;
    int* out = idx + (size_t)gid * K;
    for (int n = 0; n < N && cnt < K; ++n) {
        float dx = P[n * 3] - cx, dy = P[n * 3 + 1] - cy, dz = P[n * 3 + 2] - cz;
        if (dx * dx + dy * dy + dz * dz <= r2) {
            if (cnt == 0) first = n;
            out[cnt++] = n;
        }
    }
    for (; cnt < K; ++cnt) out[cnt] = first;
}

// sa1 grouping: rows (b,s,k) x 32 chans: [dx,dy,dz, 0-pad]
__global__ void k_group_xyz(const float* __restrict__ xyz, const float* __restrict__ nxyz,
                            const int* __restrict__ idx, int B_, int S, int K, int N,
                            f16* __restrict__ X)
{
    int gid = blockIdx.x * blockDim.x + threadIdx.x;
    if (gid >= B_ * S * K) return;
    int bs = gid / K;
    int b  = bs / S;
    int n  = idx[gid];
    f16* row = X + (size_t)gid * 32;
    row[0] = (f16)(xyz[((size_t)b * N + n) * 3 + 0] - nxyz[bs * 3 + 0]);
    row[1] = (f16)(xyz[((size_t)b * N + n) * 3 + 1] - nxyz[bs * 3 + 1]);
    row[2] = (f16)(xyz[((size_t)b * N + n) * 3 + 2] - nxyz[bs * 3 + 2]);
    for (int c = 3; c < 32; ++c) row[c] = (f16)0.f;
}

// sa2 grouping: [320 gathered feats, dx,dy,dz, 0-pad to 352]
__global__ void k_group_feat(const float* __restrict__ xyz, const float* __restrict__ nxyz,
                             const int* __restrict__ idx, const f16* __restrict__ pts,
                             int Cpts, int B_, int S, int K, int N, int Cpad,
                             f16* __restrict__ X)
{
    int gid = blockIdx.x * blockDim.x + threadIdx.x;
    if (gid >= B_ * S * K) return;
    int bs = gid / K;
    int b  = bs / S;
    int n  = idx[gid];
    const f16* src = pts + ((size_t)b * N + n) * Cpts;
    f16* row = X + (size_t)gid * Cpad;
    for (int c = 0; c < Cpts; ++c) row[c] = src[c];
    row[Cpts + 0] = (f16)(xyz[((size_t)b * N + n) * 3 + 0] - nxyz[bs * 3 + 0]);
    row[Cpts + 1] = (f16)(xyz[((size_t)b * N + n) * 3 + 1] - nxyz[bs * 3 + 1]);
    row[Cpts + 2] = (f16)(xyz[((size_t)b * N + n) * 3 + 2] - nxyz[bs * 3 + 2]);
    for (int c = Cpts + 3; c < Cpad; ++c) row[c] = (f16)0.f;
}

// pooled f32 feats -> f16 concat destination at channel offset
__global__ void k_copy_feat(const float* __restrict__ pool, int rows, int Cb,
                            f16* __restrict__ dst, int Cdst, int coff)
{
    int gid = blockIdx.x * blockDim.x + threadIdx.x;
    if (gid >= rows * Cb) return;
    int r = gid / Cb, c = gid % Cb;
    dst[(size_t)r * Cdst + coff + c] = (f16)pool[gid];
}

// sa3 input rows: [xyz(3), l2_pts(640), 0-pad to 672]
__global__ void k_build_sa3(const float* __restrict__ xyz, const f16* __restrict__ pts,
                            int rows, f16* __restrict__ X)
{
    int gid = blockIdx.x * blockDim.x + threadIdx.x;
    if (gid >= rows) return;
    f16* row = X + (size_t)gid * 672;
    row[0] = (f16)xyz[gid * 3 + 0];
    row[1] = (f16)xyz[gid * 3 + 1];
    row[2] = (f16)xyz[gid * 3 + 2];
    const f16* p = pts + (size_t)gid * 640;
    for (int c = 0; c < 640; ++c) row[3 + c] = p[c];
    for (int c = 643; c < 672; ++c) row[c] = (f16)0.f;
}

// name input rows: [site_name(20), 0-pad to 32]
__global__ void k_build_name(const float* __restrict__ sn, int rows, f16* __restrict__ X)
{
    int gid = blockIdx.x * blockDim.x + threadIdx.x;
    if (gid >= rows) return;
    const float* s = sn + (size_t)gid * 20;
    f16* row = X + (size_t)gid * 32;
    for (int c = 0; c < 20; ++c) row[c] = (f16)s[c];
    for (int c = 20; c < 32; ++c) row[c] = (f16)0.f;
}

__global__ void k_fc1(const float* __restrict__ coord, const float* __restrict__ namef,
                      const float* __restrict__ w, const float* __restrict__ bias,
                      float* __restrict__ h)
{
    int gid = blockIdx.x * blockDim.x + threadIdx.x;
    if (gid >= BB * 512) return;
    int b = gid / 512, o = gid % 512;
    const float* wr = w + (size_t)o * 1280;
    float acc = bias[o];
    const float* cf = coord + b * 1024;
    for (int c = 0; c < 1024; ++c) acc += cf[c] * wr[c];
    const float* nf = namef + b * 256;
    for (int c = 0; c < 256; ++c) acc += nf[c] * wr[1024 + c];
    h[gid] = fmaxf(acc, 0.f);
}

__global__ void k_fc2(const float* __restrict__ h, const float* __restrict__ w,
                      const float* __restrict__ bias, float* __restrict__ out)
{
    int b = blockIdx.x * blockDim.x + threadIdx.x;
    if (b >= BB) return;
    float acc = bias[0];
    const float* hb = h + b * 512;
    for (int c = 0; c < 512; ++c) acc += hb[c] * w[c];
    out[b] = 1.f / (1.f + __expf(-acc));
}

// ---------------------------------------------------------------------------
// Host-side layer driver
// ---------------------------------------------------------------------------
struct Layer { const float *w, *g, *be; int Cin, Cout; };

static inline int pad32(int c) { return (c + 31) & ~31; }

static void run_layer(const f16* X, int M, const Layer& L, f16* wbuf, float* stats,
                      f16* xnext, float* pool, int lg2G, hipStream_t st)
{
    int Kpad = pad32(L.Cin);
    int nw = L.Cout * Kpad;
    k_convert_w<<<(nw + 255) / 256, 256, 0, st>>>(L.w, L.Cin, Kpad, nw, wbuf);
    k_zero_f32 <<<(2048 + 255) / 256, 256, 0, st>>>(stats, 2048);
    int Mt2 = M / 32, Nt2 = L.Cout / 32;          // 32x32 wave blocks
    int tiles = Mt2 * Nt2;
    int blocks = (tiles + 7) / 8;                 // 8 waves (blocks) per 256-thread CTA
    k_gemm_stats<<<blocks, 256, 0, st>>>(X, wbuf, Kpad, Mt2, Nt2, stats, stats + 1024);
    k_bn_finalize<<<(L.Cout + 255) / 256, 256, 0, st>>>(L.g, L.be, stats, L.Cout,
                                                        1.0f / (float)M);
    k_gemm_apply<<<blocks, 256, 0, st>>>(X, wbuf, Kpad, Mt2, Nt2, stats + 2048,
                                         stats + 3072, xnext, pool, lg2G, L.Cout);
}

extern "C" void kernel_launch(void* const* d_in, const int* in_sizes, int n_in,
                              void* d_out, int out_size, void* d_ws, size_t ws_size,
                              hipStream_t stream)
{
    (void)in_sizes; (void)n_in; (void)out_size; (void)ws_size;
    auto F = [&](int i) { return (const float*)d_in[i]; };

    const float* site_coord = F(0);
    const float* site_name  = F(1);
    const float* fc1_b = F(2); const float* fc1_w = F(3);
    const float* fc2_b = F(4); const float* fc2_w = F(5);

    Layer nameL[2] = { { F(9),  F(8),  F(7),  20, 64  },
                       { F(13), F(12), F(11), 64, 256 } };

    static const int s1ci[3][3] = {{3,32,32},{3,64,64},{3,64,96}};
    static const int s1co[3][3] = {{32,32,64},{64,64,128},{64,96,128}};
    static const int s2ci[3][3] = {{323,64,64},{323,128,128},{323,128,128}};
    static const int s2co[3][3] = {{64,64,128},{128,128,256},{128,128,256}};
    Layer sa1L[3][3], sa2L[3][3], sa3L[3];
    for (int j = 0; j < 3; ++j)
        for (int l = 0; l < 3; ++l) {
            int b1 = 14 + (j * 3 + l) * 4;
            sa1L[j][l] = { F(b1 + 3), F(b1 + 2), F(b1 + 1), s1ci[j][l], s1co[j][l] };
            int b2 = 50 + (j * 3 + l) * 4;
            sa2L[j][l] = { F(b2 + 3), F(b2 + 2), F(b2 + 1), s2ci[j][l], s2co[j][l] };
        }
    static const int s3ci[3] = {643, 256, 512};
    static const int s3co[3] = {256, 512, 1024};
    for (int l = 0; l < 3; ++l) {
        int b3 = 86 + l * 4;
        sa3L[l] = { F(b3 + 3), F(b3 + 2), F(b3 + 1), s3ci[l], s3co[l] };
    }

    // ------------------------- workspace carve -------------------------
    char* base = (char*)d_ws; size_t off = 0;
    auto carve = [&](size_t bytes) -> void* {
        void* p = base + off; off = (off + bytes + 255) & ~(size_t)255; return p;
    };
    f16*   XA    = (f16*)  carve((size_t)131072 * 64 * sizeof(f16));
    f16*   XB    = (f16*)  carve((size_t)131072 * 64 * sizeof(f16));
    f16*   wbuf  = (f16*)  carve((size_t)1024 * 672 * sizeof(f16));
    float* stats = (float*)carve(4096 * sizeof(float));
    int*   idxf1 = (int*)  carve(BB * 128 * sizeof(int));
    float* nx1   = (float*)carve(BB * 128 * 3 * sizeof(float));
    int*   idxf2 = (int*)  carve(BB * 64 * sizeof(int));
    float* nx2   = (float*)carve(BB * 64 * 3 * sizeof(float));
    int*   idxb  = (int*)  carve(BB * 128 * 16 * sizeof(int));
    float* poolf = (float*)carve((size_t)BB * 128 * 256 * sizeof(float));
    f16*   l1pts = (f16*)  carve((size_t)BB * 128 * 320 * sizeof(f16));
    f16*   l2pts = (f16*)  carve((size_t)BB * 64 * 640 * sizeof(f16));
    float* coordf= (float*)carve(BB * 1024 * sizeof(float));
    float* namef = (float*)carve(BB * 256 * sizeof(float));
    float* hbuf  = (float*)carve(BB * 512 * sizeof(float));

    static const int   Ks[3]   = {4, 8, 16};
    static const int   lg2K[3] = {2, 3, 4};
    static const float r1[3]   = {0.1f, 0.2f, 0.4f};
    static const float r2_[3]  = {0.2f, 0.4f, 0.8f};

    // ------------------------------ sa1 ------------------------------
    k_fps<<<BB, 256, 0, stream>>>(site_coord, NPTS, 128, idxf1, nx1);
    {
        int S = 128, N = NPTS, coff = 0;
        for (int j = 0; j < 3; ++j) {
            int K = Ks[j], BS = BB * S, M = BS * K;
            k_ball<<<(BS + 127) / 128, 128, 0, stream>>>(site_coord, nx1, BB, N, S, K,
                                                         r1[j] * r1[j], idxb);
            k_group_xyz<<<(M + 255) / 256, 256, 0, stream>>>(site_coord, nx1, idxb,
                                                             BB, S, K, N, XA);
            run_layer(XA, M, sa1L[j][0], wbuf, stats, XB, nullptr, 0, stream);
            run_layer(XB, M, sa1L[j][1], wbuf, stats, XA, nullptr, 0, stream);
            int Cb = sa1L[j][2].Cout;
            k_zero_f32<<<(BS * Cb + 255) / 256, 256, 0, stream>>>(poolf, BS * Cb);
            run_layer(XA, M, sa1L[j][2], wbuf, stats, nullptr, poolf, lg2K[j], stream);
            k_copy_feat<<<(BS * Cb + 255) / 256, 256, 0, stream>>>(poolf, BS, Cb,
                                                                   l1pts, 320, coff);
            coff += Cb;
        }
    }

    // ------------------------------ sa2 ------------------------------
    k_fps<<<BB, 256, 0, stream>>>(nx1, 128, 64, idxf2, nx2);
    {
        int S = 64, N = 128, coff = 0;
        for (int j = 0; j < 3; ++j) {
            int K = Ks[j], BS = BB * S, M = BS * K;
            k_ball<<<(BS + 127) / 128, 128, 0, stream>>>(nx1, nx2, BB, N, S, K,
                                                         r2_[j] * r2_[j], idxb);
            k_group_feat<<<(M + 255) / 256, 256, 0, stream>>>(nx1, nx2, idxb, l1pts,
                                                              320, BB, S, K, N, 352, XA);
            run_layer(XA, M, sa2L[j][0], wbuf, stats, XB, nullptr, 0, stream);
            run_layer(XB, M, sa2L[j][1], wbuf, stats, XA, nullptr, 0, stream);
            int Cb = sa2L[j][2].Cout;
            k_zero_f32<<<(BS * Cb + 255) / 256, 256, 0, stream>>>(poolf, BS * Cb);
            run_layer(XA, M, sa2L[j][2], wbuf, stats, nullptr, poolf, lg2K[j], stream);
            k_copy_feat<<<(BS * Cb + 255) / 256, 256, 0, stream>>>(poolf, BS, Cb,
                                                                   l2pts, 640, coff);
            coff += Cb;
        }
    }

    // ------------------------------ sa3 ------------------------------
    {
        int rows = BB * 64;
        k_build_sa3<<<(rows + 255) / 256, 256, 0, stream>>>(nx2, l2pts, rows, XA);
        run_layer(XA, rows, sa3L[0], wbuf, stats, XB, nullptr, 0, stream);
        run_layer(XB, rows, sa3L[1], wbuf, stats, XA, nullptr, 0, stream);
        k_zero_f32<<<(BB * 1024 + 255) / 256, 256, 0, stream>>>(coordf, BB * 1024);
        run_layer(XA, rows, sa3L[2], wbuf, stats, nullptr, coordf, 6, stream);
    }

    // --------------------------- name branch ---------------------------
    {
        int rows = BB * NPTS;
        k_build_name<<<(rows + 255) / 256, 256, 0, stream>>>(site_name, rows, XA);
        run_layer(XA, rows, nameL[0], wbuf, stats, XB, nullptr, 0, stream);
        k_zero_f32<<<(BB * 256 + 255) / 256, 256, 0, stream>>>(namef, BB * 256);
        run_layer(XB, rows, nameL[1], wbuf, stats, nullptr, namef, 13, stream);
    }

    // ------------------------------- FCs -------------------------------
    k_fc1<<<(BB * 512 + 255) / 256, 256, 0, stream>>>(coordf, namef, fc1_w, fc1_b, hbuf);
    k_fc2<<<1, 32, 0, stream>>>(hbuf, fc2_w, fc2_b, (float*)d_out);
}
